// GR_14757507629551
// MI455X (gfx1250) — matrix operand order repaired and checked by reading, not compiled, not run on hardware
//
#include <hip/hip_runtime.h>
#include <hip/hip_bf16.h>

// ---------------------------------------------------------------------------
// Problem constants (from the reference)
// ---------------------------------------------------------------------------
#define B_ 8
#define W_ 2048
#define C_ 512
#define M_ 128

typedef __attribute__((ext_vector_type(16))) __bf16 v16bf;
typedef __attribute__((ext_vector_type(8)))  float  v8f;
typedef __attribute__((ext_vector_type(4)))  int    v4i;

// GEMM tiling: 128 threads (4 waves, 2x2), block tile 64x64, Kc = 32.
// Each wave computes a 32x32 sub-tile = 4 WMMAs per K stage.
#define TKC 32
#define LDK 40      // padded LDS row stride (bf16 elems); 80 B rows, 16B-aligned
#define LDS_DEG 136

// ---------------------------------------------------------------------------
// Async global->LDS DMA (GLOBAL_LOAD_ASYNC_TO_LDS_B128, ASYNCcnt), guarded so
// toolchains without the builtin fall back to the VGPR round-trip copy.
// Builtin signature (from hipcc diagnostic): (v4i AS1*, v4i AS3*, imm, imm).
// ---------------------------------------------------------------------------
#if defined(__has_builtin)
#if __has_builtin(__builtin_amdgcn_global_load_async_to_lds_b128)
#define GR_HAVE_ASYNC_LDS 1
#endif
#endif

#if defined(GR_HAVE_ASYNC_LDS)
typedef __attribute__((address_space(1))) v4i* gr_gptr_b128;
typedef __attribute__((address_space(3))) v4i* gr_lptr_b128;

__device__ __forceinline__ void gr_async_b128(const void* g, void* l) {
  __builtin_amdgcn_global_load_async_to_lds_b128((gr_gptr_b128)g,
                                                 (gr_lptr_b128)l, 0, 0);
}
__device__ __forceinline__ void gr_wait_async() {
#if __has_builtin(__builtin_amdgcn_s_wait_asynccnt)
  __builtin_amdgcn_s_wait_asynccnt(0);
#else
  asm volatile("s_wait_asynccnt 0" ::: "memory");
#endif
}
#endif

// Copy 8 contiguous bf16 (16 B) global -> LDS, async when available.
__device__ __forceinline__ void gr_copy16(const __bf16* __restrict__ g,
                                          __bf16* __restrict__ l) {
#if defined(GR_HAVE_ASYNC_LDS)
  gr_async_b128((const void*)g, (void*)l);
#else
#pragma unroll
  for (int j = 0; j < 8; ++j) l[j] = g[j];
#endif
}

// Fence staging writes (async + ds) before the workgroup barrier.
__device__ __forceinline__ void gr_stage_fence() {
#if defined(GR_HAVE_ASYNC_LDS)
  gr_wait_async();
#endif
  __syncthreads();
}

__device__ __forceinline__ float gr_sigmoid(float v) {
  return 1.0f / (1.0f + __expf(-v));
}

__device__ __forceinline__ v8f wmma_bf16(v16bf a, v16bf b, v8f c) {
  // (neg_a, A, neg_b, B, c_mod, C, reuse_a, reuse_b)
  return __builtin_amdgcn_wmma_f32_16x16x32_bf16(false, a, false, b, (short)0,
                                                 c, false, false);
}

// A fragment: 16x32 bf16 from LDS tile stored [row][k], stride LDK.
// Per lane: 16 contiguous elements (32 B) -> 2x ds_load_b128.
__device__ __forceinline__ v16bf load_frag_a(const __bf16* __restrict__ base,
                                             int lane) {
  const int row = lane & 15;
  const int kh  = lane >> 4;
  v16bf a;
#pragma unroll
  for (int e = 0; e < 16; ++e) {
    const int k = (kh << 4) + e;
    a[e] = base[row * LDK + k];
  }
  return a;
}

// B fragment: 32x16 bf16 from LDS tile stored [n][k] (B^T), stride LDK.
// Per lane: 16 contiguous elements (32 B) -> 2x ds_load_b128.
__device__ __forceinline__ v16bf load_frag_bT(const __bf16* __restrict__ base,
                                              int lane) {
  const int col = lane & 15;
  const int kh  = lane >> 4;
  v16bf b;
#pragma unroll
  for (int e = 0; e < 16; ++e) b[e] = base[col * LDK + (kh << 4) + e];
  return b;
}

// B fragment from a [n][k] tile with arbitrary stride (deg kernel).
__device__ __forceinline__ v16bf load_frag_b_nk(const __bf16* __restrict__ base,
                                                int lane, int ld, int k0) {
  const int col = lane & 15;
  const int kh  = lane >> 4;
  v16bf b;
#pragma unroll
  for (int e = 0; e < 16; ++e) b[e] = base[col * ld + k0 + (kh << 4) + e];
  return b;
}

// ---------------------------------------------------------------------------
// Small helper kernels
// ---------------------------------------------------------------------------
__global__ void gr_cvt_bf16(const float* __restrict__ in,
                            __bf16* __restrict__ out, int n) {
  for (int i = blockIdx.x * blockDim.x + threadIdx.x; i < n;
       i += gridDim.x * blockDim.x)
    out[i] = (__bf16)in[i];
}

__global__ void gr_reduce(const float* __restrict__ x,
                          float* __restrict__ xavg,
                          float* __restrict__ xmax) {
  const int idx = blockIdx.x * blockDim.x + threadIdx.x;  // 0 .. B*C-1
  const int b = idx / C_;
  const int c = idx - b * C_;
  float s = 0.0f, mx = -3.4e38f;
  for (int w = 0; w < W_; ++w) {
    const float v = x[(size_t)(b * W_ + w) * C_ + c];
    s += v;
    mx = fmaxf(mx, v);
  }
  xavg[idx] = s * (1.0f / (float)W_);
  xmax[idx] = mx;
}

__global__ void gr_cw(const float* __restrict__ xavg,
                      const float* __restrict__ xmax,
                      const float* __restrict__ avg_w,
                      const float* __restrict__ max_w,
                      float* __restrict__ cw) {
  const int idx = blockIdx.x * blockDim.x + threadIdx.x;  // 0 .. B*M-1
  const int b = idx / M_;
  const int m = idx - b * M_;
  float a = 0.0f, mm = 0.0f;
  for (int c = 0; c < C_; ++c) {
    a  += xavg[b * C_ + c] * avg_w[m * C_ + c];
    mm += xmax[b * C_ + c] * max_w[m * C_ + c];
  }
  a  = fmaxf(a, 0.0f);
  mm = fmaxf(mm, 0.0f);
  cw[idx] = gr_sigmoid(a + mm);
}

__global__ void gr_scale_p(const float* __restrict__ xxf,
                           const float* __restrict__ dvec,
                           __bf16* __restrict__ pbf, int n) {
  for (int i = blockIdx.x * blockDim.x + threadIdx.x; i < n;
       i += gridDim.x * blockDim.x)
    pbf[i] = (__bf16)(xxf[i] * dvec[i / M_]);
}

// ---------------------------------------------------------------------------
// Shared GEMM per-stage compute: 4 WMMAs on a 32x32 wave tile.
// ---------------------------------------------------------------------------
__device__ __forceinline__ void gemm_stage(const __bf16* As, const __bf16* Bs,
                                           int wr, int wc, int lane,
                                           v8f acc[2][2]) {
  const v16bf a0 = load_frag_a(As + (wr * 32) * LDK, lane);
  const v16bf a1 = load_frag_a(As + (wr * 32 + 16) * LDK, lane);
  const v16bf b0 = load_frag_bT(Bs + (wc * 32) * LDK, lane);
  const v16bf b1 = load_frag_bT(Bs + (wc * 32 + 16) * LDK, lane);
  acc[0][0] = wmma_bf16(a0, b0, acc[0][0]);
  acc[0][1] = wmma_bf16(a0, b1, acc[0][1]);
  acc[1][0] = wmma_bf16(a1, b0, acc[1][0]);
  acc[1][1] = wmma_bf16(a1, b1, acc[1][1]);
}

// ---------------------------------------------------------------------------
// GEMM 1: xx[b,w,m] = x[b,w,:] . fc_w[m,:] + fc_b[m]; dq = xx * cw
// A = x_bf [W x C] natural; B = fc_w [M x C] -> already [n][k], direct copy.
// ---------------------------------------------------------------------------
__global__ __launch_bounds__(128) void gr_gemm_xx(
    const __bf16* __restrict__ xbf, const __bf16* __restrict__ fcwbf,
    const float* __restrict__ fc_b, const float* __restrict__ cw,
    float* __restrict__ xxf, __bf16* __restrict__ xxbf,
    __bf16* __restrict__ dqbf) {
  __shared__ __bf16 As[64 * LDK];
  __shared__ __bf16 Bs[64 * LDK];
  const int b   = blockIdx.z;
  const int w0  = blockIdx.x * 64;
  const int m0  = blockIdx.y * 64;
  const int tid = threadIdx.x;
  const int wv = tid >> 5, lane = tid & 31;
  const int wr = wv >> 1, wc = wv & 1;

  v8f acc[2][2] = {};
  for (int kb = 0; kb < C_ / TKC; ++kb) {
    const int c0 = kb * TKC;
    {  // A: two rows of 8 contiguous bf16 per thread (16 B chunks)
      const int r = tid >> 2, k8 = (tid & 3) << 3;
      if (kb + 1 < C_ / TKC)
        __builtin_prefetch(&xbf[(size_t)(b * W_ + w0 + r) * C_ + c0 + TKC], 0, 1);
#pragma unroll
      for (int h = 0; h < 2; ++h) {
        const int row = r + h * 32;
        gr_copy16(&xbf[(size_t)(b * W_ + w0 + row) * C_ + c0 + k8],
                  &As[row * LDK + k8]);
      }
    }
    {  // B: fc_w is [m][c] = [n][k]; two 16 B chunks per thread, direct copy
      const int n = tid & 63, kq = (tid >> 6) << 4;
#pragma unroll
      for (int h = 0; h < 2; ++h)
        gr_copy16(&fcwbf[(size_t)(m0 + n) * C_ + c0 + kq + h * 8],
                  &Bs[n * LDK + kq + h * 8]);
    }
    gr_stage_fence();
    gemm_stage(As, Bs, wr, wc, lane, acc);
    __syncthreads();
  }

  const int kh = lane >> 4, col = lane & 15;
#pragma unroll
  for (int fc = 0; fc < 2; ++fc) {
    const int m = m0 + wc * 32 + fc * 16 + col;
    const float cwv  = cw[b * M_ + m];
    const float bias = fc_b[m];
#pragma unroll
    for (int fr = 0; fr < 2; ++fr) {
#pragma unroll
      for (int r = 0; r < 8; ++r) {
        const int w = w0 + wr * 32 + fr * 16 + (kh << 3) + r;
        const float v = acc[fr][fc][r] + bias;
        const size_t idx = (size_t)(b * W_ + w) * M_ + m;
        xxf[idx]  = v;
        xxbf[idx] = (__bf16)v;
        dqbf[idx] = (__bf16)(v * cwv);
      }
    }
  }
}

// ---------------------------------------------------------------------------
// Fused deg kernel: deg[b,w] = sum_v sigmoid(dq[b,w,:].xx[b,v,:]); d=deg^-1/2
// ---------------------------------------------------------------------------
__global__ __launch_bounds__(256) void gr_deg(
    const __bf16* __restrict__ dqbf, const __bf16* __restrict__ xxbf,
    float* __restrict__ dvec) {
  __shared__ __bf16 xs[16 * LDS_DEG];
  const int b    = blockIdx.y;
  const int wv   = threadIdx.x >> 5;
  const int lane = threadIdx.x & 31;
  const int wbase = blockIdx.x * 128 + wv * 16;
  const int row = lane & 15, kh = lane >> 4, col = lane & 15;

  // Register-resident A fragments: this wave's 16 rows x 128 K of dq.
  v16bf afr[4];
#pragma unroll
  for (int ks = 0; ks < 4; ++ks) {
#pragma unroll
    for (int e = 0; e < 16; ++e) {
      const int k = ks * 32 + (kh << 4) + e;
      afr[ks][e] = dqbf[(size_t)(b * W_ + wbase + row) * M_ + k];
    }
  }

  float rs[8];
#pragma unroll
  for (int r = 0; r < 8; ++r) rs[r] = 0.0f;

  const int vr = threadIdx.x >> 4, m8 = (threadIdx.x & 15) << 3;
  for (int vt = 0; vt < W_ / 16; ++vt) {
    const int v0 = vt * 16;
    // stage xx tile [16 v][128 m] into LDS ([n][k] layout), 16 B per thread
    gr_copy16(&xxbf[(size_t)(b * W_ + v0 + vr) * M_ + m8],
              &xs[vr * LDS_DEG + m8]);
    gr_stage_fence();
    v8f acc = {};
#pragma unroll
    for (int ks = 0; ks < 4; ++ks) {
      const v16bf bb = load_frag_b_nk(xs, lane, LDS_DEG, ks * 32);
      acc = wmma_bf16(afr[ks], bb, acc);
    }
#pragma unroll
    for (int r = 0; r < 8; ++r) rs[r] += gr_sigmoid(acc[r]);
    __syncthreads();
  }

#pragma unroll
  for (int mask = 1; mask < 16; mask <<= 1) {
#pragma unroll
    for (int r = 0; r < 8; ++r) rs[r] += __shfl_xor(rs[r], mask, 32);
  }
  if (col == 0) {
#pragma unroll
    for (int r = 0; r < 8; ++r)
      dvec[b * W_ + wbase + (kh << 3) + r] = rsqrtf(rs[r]);
  }
}

// ---------------------------------------------------------------------------
// GEMM 2: LXs[b,m,c] = cw[b,m] * sum_w P[b,w,m] * x[b,w,c]   (K = W)
// A = P^T (transpose-staged), B = x_bf [k][n] (transpose-staged to [n][k]).
// ---------------------------------------------------------------------------
__global__ __launch_bounds__(128) void gr_gemm_lx(
    const __bf16* __restrict__ pbf, const __bf16* __restrict__ xbf,
    const float* __restrict__ cw, __bf16* __restrict__ lxsbf) {
  __shared__ __bf16 As[64 * LDK];
  __shared__ __bf16 Bs[64 * LDK];
  const int b   = blockIdx.z;
  const int m0  = blockIdx.x * 64;
  const int c0  = blockIdx.y * 64;
  const int tid = threadIdx.x;
  const int wv = tid >> 5, lane = tid & 31;
  const int wr = wv >> 1, wc = wv & 1;

  v8f acc[2][2] = {};
  const int n = tid & 63, kq = (tid >> 6) << 4;
  for (int kb = 0; kb < W_ / TKC; ++kb) {
    const int w0 = kb * TKC;
    if (kb + 1 < W_ / TKC)
      __builtin_prefetch(&pbf[(size_t)(b * W_ + w0 + TKC + kq) * M_ + m0 + n], 0, 1);
#pragma unroll
    for (int j = 0; j < 16; ++j) {  // As[m][k=w] = P[w0+k][m0+m]
      const int k = kq + j;
      As[n * LDK + k] = pbf[(size_t)(b * W_ + w0 + k) * M_ + m0 + n];
    }
#pragma unroll
    for (int j = 0; j < 16; ++j) {  // Bs[n=c][k=w] = x[w0+k][c0+n]
      const int k = kq + j;
      Bs[n * LDK + k] = xbf[(size_t)(b * W_ + w0 + k) * C_ + c0 + n];
    }
    __syncthreads();
    gemm_stage(As, Bs, wr, wc, lane, acc);
    __syncthreads();
  }

  const int kh = lane >> 4, col = lane & 15;
#pragma unroll
  for (int fc = 0; fc < 2; ++fc) {
    const int c = c0 + wc * 32 + fc * 16 + col;
#pragma unroll
    for (int fr = 0; fr < 2; ++fr) {
#pragma unroll
      for (int r = 0; r < 8; ++r) {
        const int m = m0 + wr * 32 + fr * 16 + (kh << 3) + r;
        lxsbf[(size_t)(b * M_ + m) * C_ + c] =
            (__bf16)(acc[fr][fc][r] * cw[b * M_ + m]);
      }
    }
  }
}

// ---------------------------------------------------------------------------
// GEMM 3: T[b,w,c] = x[b,w,c] - sum_m P[b,w,m] * LXs[b,m,c]   (K = M = 128)
// A = P natural (async-staged); B = LXs [k][n] (transpose-staged).
// ---------------------------------------------------------------------------
__global__ __launch_bounds__(128) void gr_gemm_y(
    const __bf16* __restrict__ pbf, const __bf16* __restrict__ lxsbf,
    const float* __restrict__ x, __bf16* __restrict__ tbf) {
  __shared__ __bf16 As[64 * LDK];
  __shared__ __bf16 Bs[64 * LDK];
  const int b   = blockIdx.z;
  const int w0  = blockIdx.x * 64;
  const int c0  = blockIdx.y * 64;
  const int tid = threadIdx.x;
  const int wv = tid >> 5, lane = tid & 31;
  const int wr = wv >> 1, wc = wv & 1;

  v8f acc[2][2] = {};
  for (int kb = 0; kb < M_ / TKC; ++kb) {
    const int k0 = kb * TKC;
    {  // A natural: P[w][m], 16 B chunks
      const int r = tid >> 2, k8 = (tid & 3) << 3;
#pragma unroll
      for (int h = 0; h < 2; ++h) {
        const int row = r + h * 32;
        gr_copy16(&pbf[(size_t)(b * W_ + w0 + row) * M_ + k0 + k8],
                  &As[row * LDK + k8]);
      }
    }
    {  // B transpose: Bs[n=c][k=m] = LXs[k0+k][c0+n]
      const int n = tid & 63, kq = (tid >> 6) << 4;
#pragma unroll
      for (int j = 0; j < 16; ++j) {
        const int k = kq + j;
        Bs[n * LDK + k] = lxsbf[(size_t)(b * M_ + k0 + k) * C_ + c0 + n];
      }
    }
    gr_stage_fence();
    gemm_stage(As, Bs, wr, wc, lane, acc);
    __syncthreads();
  }

  const int kh = lane >> 4, col = lane & 15;
#pragma unroll
  for (int fc = 0; fc < 2; ++fc) {
    const int c = c0 + wc * 32 + fc * 16 + col;
#pragma unroll
    for (int fr = 0; fr < 2; ++fr) {
#pragma unroll
      for (int r = 0; r < 8; ++r) {
        const int w = w0 + wr * 32 + fr * 16 + (kh << 3) + r;
        const size_t idx = (size_t)(b * W_ + w) * C_ + c;
        tbf[idx] = (__bf16)(x[idx] - acc[fr][fc][r]);
      }
    }
  }
}

// ---------------------------------------------------------------------------
// GEMM 4: out[b,w,c] = sum_k T[b,w,k] * gcn_w[k,c]   (K = C = 512)
// A = T natural (async-staged); B = gcn_w [k][n] (transpose-staged).
// ---------------------------------------------------------------------------
__global__ __launch_bounds__(128) void gr_gemm_out(
    const __bf16* __restrict__ tbf, const __bf16* __restrict__ gcnwbf,
    float* __restrict__ out) {
  __shared__ __bf16 As[64 * LDK];
  __shared__ __bf16 Bs[64 * LDK];
  const int b   = blockIdx.z;
  const int w0  = blockIdx.x * 64;
  const int c0  = blockIdx.y * 64;
  const int tid = threadIdx.x;
  const int wv = tid >> 5, lane = tid & 31;
  const int wr = wv >> 1, wc = wv & 1;

  v8f acc[2][2] = {};
  for (int kb = 0; kb < C_ / TKC; ++kb) {
    const int k0 = kb * TKC;
    {
      const int r = tid >> 2, k8 = (tid & 3) << 3;
      if (kb + 1 < C_ / TKC)
        __builtin_prefetch(&tbf[(size_t)(b * W_ + w0 + r) * C_ + k0 + TKC], 0, 1);
#pragma unroll
      for (int h = 0; h < 2; ++h) {
        const int row = r + h * 32;
        gr_copy16(&tbf[(size_t)(b * W_ + w0 + row) * C_ + k0 + k8],
                  &As[row * LDK + k8]);
      }
    }
    {
      const int n = tid & 63, kq = (tid >> 6) << 4;
#pragma unroll
      for (int j = 0; j < 16; ++j) {
        const int k = kq + j;
        Bs[n * LDK + k] = gcnwbf[(size_t)(k0 + k) * C_ + c0 + n];
      }
    }
    gr_stage_fence();
    gemm_stage(As, Bs, wr, wc, lane, acc);
    __syncthreads();
  }

  const int kh = lane >> 4, col = lane & 15;
#pragma unroll
  for (int fc = 0; fc < 2; ++fc) {
    const int c = c0 + wc * 32 + fc * 16 + col;
#pragma unroll
    for (int fr = 0; fr < 2; ++fr) {
#pragma unroll
      for (int r = 0; r < 8; ++r) {
        const int w = w0 + wr * 32 + fr * 16 + (kh << 3) + r;
        out[(size_t)(b * W_ + w) * C_ + c] = acc[fr][fc][r];
      }
    }
  }
}

// ---------------------------------------------------------------------------
// Launch
// ---------------------------------------------------------------------------
extern "C" void kernel_launch(void* const* d_in, const int* in_sizes, int n_in,
                              void* d_out, int out_size, void* d_ws,
                              size_t ws_size, hipStream_t stream) {
  const float* x     = (const float*)d_in[0];
  const float* fc_w  = (const float*)d_in[1];
  const float* fc_b  = (const float*)d_in[2];
  const float* avg_w = (const float*)d_in[3];
  const float* max_w = (const float*)d_in[4];
  const float* gcn_w = (const float*)d_in[5];
  float* out = (float*)d_out;

  char* base = (char*)d_ws;
  size_t off = 0;
  auto take = [&](size_t bytes) -> void* {
    void* q = base + off;
    off = (off + bytes + 255) & ~(size_t)255;
    return q;
  };

  __bf16* x_bf    = (__bf16*)take((size_t)B_ * W_ * C_ * 2);
  __bf16* fcw_bf  = (__bf16*)take((size_t)M_ * C_ * 2);
  __bf16* gcnw_bf = (__bf16*)take((size_t)C_ * C_ * 2);
  float*  xavg    = (float*)take((size_t)B_ * C_ * 4);
  float*  xmax    = (float*)take((size_t)B_ * C_ * 4);
  float*  cw      = (float*)take((size_t)B_ * M_ * 4);
  float*  xx_f    = (float*)take((size_t)B_ * W_ * M_ * 4);
  __bf16* xx_bf   = (__bf16*)take((size_t)B_ * W_ * M_ * 2);
  __bf16* dq_bf   = (__bf16*)take((size_t)B_ * W_ * M_ * 2);
  float*  dvec    = (float*)take((size_t)B_ * W_ * 4);
  __bf16* p_bf    = (__bf16*)take((size_t)B_ * W_ * M_ * 2);
  __bf16* lxs_bf  = (__bf16*)take((size_t)B_ * M_ * C_ * 2);
  __bf16* t_bf    = (__bf16*)take((size_t)B_ * W_ * C_ * 2);
  (void)ws_size; (void)in_sizes; (void)n_in; (void)out_size;

  gr_cvt_bf16<<<4096, 256, 0, stream>>>(x, x_bf, B_ * W_ * C_);
  gr_cvt_bf16<<<256, 256, 0, stream>>>(fc_w, fcw_bf, M_ * C_);
  gr_cvt_bf16<<<1024, 256, 0, stream>>>(gcn_w, gcnw_bf, C_ * C_);
  gr_reduce<<<(B_ * C_) / 256, 256, 0, stream>>>(x, xavg, xmax);
  gr_cw<<<(B_ * M_) / 256, 256, 0, stream>>>(xavg, xmax, avg_w, max_w, cw);

  gr_gemm_xx<<<dim3(W_ / 64, M_ / 64, B_), 128, 0, stream>>>(
      x_bf, fcw_bf, fc_b, cw, xx_f, xx_bf, dq_bf);

  gr_deg<<<dim3(W_ / 128, B_), 256, 0, stream>>>(dq_bf, xx_bf, dvec);

  gr_scale_p<<<2048, 256, 0, stream>>>(xx_f, dvec, p_bf, B_ * W_ * M_);

  gr_gemm_lx<<<dim3(M_ / 64, C_ / 64, B_), 128, 0, stream>>>(
      p_bf, x_bf, cw, lxs_bf);

  gr_gemm_y<<<dim3(W_ / 64, C_ / 64, B_), 128, 0, stream>>>(
      p_bf, lxs_bf, x, t_bf);

  gr_gemm_out<<<dim3(W_ / 64, C_ / 64, B_), 128, 0, stream>>>(
      t_bf, gcnw_bf, out);
}
